// Wave2D_42752104464835
// MI455X (gfx1250) — matrix operand order repaired and checked by reading, not compiled, not run on hardware
//
#include <hip/hip_runtime.h>
#include <hip/hip_bf16.h>
#include <math.h>

#define DEVI __device__ __forceinline__

typedef __bf16 bf16;
typedef __attribute__((ext_vector_type(16))) __bf16 v16bf;
typedef __attribute__((ext_vector_type(8)))  float   v8f;

static constexpr int B_  = 32;
static constexpr int C_  = 384;
static constexpr int H_  = 64;
static constexpr int W_  = 64;
static constexpr int HW_ = H_ * W_;           // 4096
static constexpr int BHW_ = B_ * HW_;         // 131072
static constexpr float PI_ = 3.14159265358979323846f;
static constexpr int PAD_ = 72;               // DCT LDS row pitch (144B, 16B aligned)

// ---- workspace layout (bytes) ----
static constexpr size_t WS_COS   = 0;                                   // bf16 [64][64]
static constexpr size_t WS_WMOD  = 8192;                                // f32  [4096][384]
static constexpr size_t WS_STATS = WS_WMOD + (size_t)HW_ * C_ * 4;      // float2 [131072]
static constexpr size_t WS_A     = WS_STATS + (size_t)BHW_ * 8;         // bf16 planar [B][C][HW]
static constexpr size_t WS_X     = WS_A + (size_t)BHW_ * C_ * 2;        // bf16 planar
static constexpr size_t WS_Z     = WS_X + (size_t)BHW_ * C_ * 2;        // bf16 planar

// ================= WMMA fragment helpers (ISA 7.12.2 layouts) =================
DEVI int kmapA(int i, int half) { return (i < 8) ? (half * 8 + i) : (16 + half * 8 + (i - 8)); }
DEVI int kmapB(int i, int half) { return (i < 8) ? (half * 8 + i) : (16 + half * 8 + (i - 8)); }

DEVI v16bf frag_a(const bf16* p, int sM, int sK) {
  const int lane = threadIdx.x & 31;
  const int m = lane & 15, h = (lane >> 4) & 1;
  v16bf a;
#pragma unroll
  for (int i = 0; i < 16; ++i) a[i] = p[m * sM + kmapA(i, h) * sK];
  return a;
}
DEVI v16bf frag_b(const bf16* p, int sK, int sN) {
  const int lane = threadIdx.x & 31;
  const int n = lane & 15, h = (lane >> 4) & 1;
  v16bf b;
#pragma unroll
  for (int i = 0; i < 16; ++i) b[i] = p[kmapB(i, h) * sK + n * sN];
  return b;
}
DEVI v8f wmma_bf16(v16bf a, v16bf b, v8f c) {
  return __builtin_amdgcn_wmma_f32_16x16x32_bf16(false, a, false, b, (short)0, c, false, false);
}
DEVI v8f vzero() {
  v8f z;
#pragma unroll
  for (int i = 0; i < 8; ++i) z[i] = 0.f;
  return z;
}
DEVI void store_tile_bf16(bf16* dst, int stride, int tm, int tn, const v8f& a) {
  const int lane = threadIdx.x & 31;
  const int hl = lane >> 4, n = lane & 15;
#pragma unroll
  for (int r = 0; r < 8; ++r)
    dst[(tm * 16 + r + hl * 8) * stride + tn * 16 + n] = (bf16)a[r];
}
// 64x64x64 matmul step for the DCT kernel (K loop of 2x32)
DEVI v8f mm64(const bf16* A, int AsM, int AsK, const bf16* Bp, int BsK, int BsN, int tm, int tn) {
  v8f acc = vzero();
#pragma unroll
  for (int kk = 0; kk < 64; kk += 32) {
    v16bf a = frag_a(A + tm * 16 * AsM + kk * AsK, AsM, AsK);
    v16bf b = frag_b(Bp + kk * BsK + tn * 16 * BsN, BsK, BsN);
    acc = wmma_bf16(a, b, acc);
  }
  return acc;
}
DEVI float gelu_tanh(float v) {
  float u = 0.7978845608f * (v + 0.044715f * v * v * v);
  return 0.5f * v * (1.f + tanhf(u));
}
// async bf16 global->LDS copy of one 16B chunk (ASYNCcnt tracked)
DEVI void async_copy16(const bf16* gsrc, const bf16* ldst) {
  unsigned lds_off = (unsigned)(uintptr_t)ldst;   // low 32 bits = LDS byte offset
  asm volatile("global_load_async_to_lds_b128 %0, %1, off"
               :: "v"(lds_off), "v"(gsrc) : "memory");
}
DEVI void wait_async0() {
  asm volatile("s_wait_asynccnt 0x0" ::: "memory");
}

// ================= K0: DCT-II orthonormal basis (bf16) =================
__global__ __launch_bounds__(256) void k_cos(bf16* __restrict__ cosB) {
  int i = blockIdx.x * 256 + threadIdx.x;
  if (i < 64 * 64) {
    int n = i >> 6, x = i & 63;
    float w = cosf(PI_ * n * ((x + 0.5f) / 64.f));
    float coef = (n == 0) ? 0.125f : 0.17677669529663687f;
    cosB[i] = (bf16)(w * coef);
  }
}

// ================= K1: wmod = exp(-lam*gelu(fe@to_k^T+b)) * tmod =================
__global__ __launch_bounds__(256) void k_wmod(const float* __restrict__ fe,
                                              const float* __restrict__ tkw,
                                              const float* __restrict__ tkb,
                                              const float* __restrict__ cs,
                                              const float* __restrict__ al,
                                              float* __restrict__ wmod) {
  const int pbase = blockIdx.x * 128;   // freq position (n*64+m)
  const int nbase = blockIdx.y * 64;    // output channel
  __shared__ bf16  sA[128 * 34];
  __shared__ bf16  sB[64 * 34];
  __shared__ float sO[64 * 130];        // f32 staging (kills accumulator spills)
  const int tid = threadIdx.x;
  const int wid = tid >> 5, lane = tid & 31, hl = lane >> 4, n16 = lane & 15;
  const int wm = (wid & 3) * 32, wn = (wid >> 2) * 32;
  v8f acc[2][2];
#pragma unroll
  for (int im = 0; im < 2; ++im)
#pragma unroll
    for (int in = 0; in < 2; ++in) acc[im][in] = vzero();

  for (int kk = 0; kk < C_; kk += 32) {
    __syncthreads();
    for (int i = tid; i < 128 * 32; i += 256) {
      int m = i >> 5, k = i & 31;
      sA[m * 34 + k] = (bf16)fe[(size_t)(pbase + m) * C_ + kk + k];
    }
    for (int i = tid; i < 64 * 32; i += 256) {
      int n = i >> 5, k = i & 31;
      sB[n * 34 + k] = (bf16)tkw[(size_t)(nbase + n) * C_ + kk + k];
    }
    __syncthreads();
#pragma unroll
    for (int im = 0; im < 2; ++im) {
      v16bf a = frag_a(&sA[(wm + im * 16) * 34], 34, 1);
#pragma unroll
      for (int in = 0; in < 2; ++in) {
        v16bf b = frag_b(&sB[(wn + in * 16) * 34], 1, 34);
        acc[im][in] = wmma_bf16(a, b, acc[im][in]);
      }
    }
  }
  // stage raw accumulators to LDS, release VGPRs before transcendental epilogue
#pragma unroll
  for (int im = 0; im < 2; ++im)
#pragma unroll
    for (int in = 0; in < 2; ++in)
#pragma unroll
      for (int r = 0; r < 8; ++r) {
        int m = wm + im * 16 + r + hl * 8;
        int n = wn + in * 16 + n16;
        sO[n * 130 + m] = acc[im][in][r];
      }
  __syncthreads();
  const float c0 = cs[0];
  const float ea = expf(-al[0]);
  for (int i = tid; i < 128 * 64; i += 256) {
    int m = i >> 6, n = i & 63;            // n fastest -> coalesced wmod writes
    int p = pbase + m, co = nbase + n;
    float kv = gelu_tanh(sO[n * 130 + m] + tkb[co]);
    int fn = p >> 6, fm = p & 63;
    float ln = fn * (PI_ / 64.f); ln *= ln;
    float lm = fm * (PI_ / 64.f); lm *= lm;
    float lam = ln + lm;
    float cw = c0 * sqrtf(lam);
    float tm_ = ea * (cosf(cw) + sinf(cw) / (cw + 1e-6f));
    wmod[(size_t)p * C_ + co] = expf(-lam * kv) * tm_;
  }
}

// ================= K2: depthwise 3x3 conv (planar) -> bf16 =================
__global__ __launch_bounds__(256) void k_dwconv(const float* __restrict__ x,
                                                const float* __restrict__ w,
                                                const float* __restrict__ bias,
                                                bf16* __restrict__ out) {
  const int plane = blockIdx.x;      // b*C + c
  const int c = plane % C_;
  __shared__ float p[64 * 64];
  const float* src = x + (size_t)plane * HW_;
  for (int i = threadIdx.x; i < HW_; i += 256) p[i] = src[i];
  float wv[9];
#pragma unroll
  for (int j = 0; j < 9; ++j) wv[j] = w[c * 9 + j];
  const float bb = bias[c];
  __syncthreads();
  bf16* dst = out + (size_t)plane * HW_;
  for (int i = threadIdx.x; i < HW_; i += 256) {
    int h = i >> 6, col = i & 63;
    float acc = bb;
#pragma unroll
    for (int dh = -1; dh <= 1; ++dh) {
      int hh = h + dh;
      if (hh < 0 || hh > 63) continue;
#pragma unroll
      for (int dw = -1; dw <= 1; ++dw) {
        int ww = col + dw;
        if (ww < 0 || ww > 63) continue;
        acc += wv[(dh + 1) * 3 + (dw + 1)] * p[hh * 64 + ww];
      }
    }
    dst[i] = (bf16)acc;
  }
}

// ================= K3: xz = xc @ lin_w^T + lin_b  (M=131072, N=768, K=384) =================
__global__ __launch_bounds__(256) void k_gemm_lin(const bf16* __restrict__ xc,
                                                  const float* __restrict__ lw,
                                                  const float* __restrict__ lb,
                                                  bf16* __restrict__ xout,
                                                  bf16* __restrict__ zout) {
  const int pbase = blockIdx.x * 128;
  const int nbase = blockIdx.y * 64;
  const int bI = pbase >> 12, hw0 = pbase & 4095;
  __shared__ bf16 sA[128 * 34];     // [m][k]
  __shared__ bf16 sB[64 * 34];      // [n][k]
  __shared__ bf16 sO[64 * 130];     // [n][m]
  const int tid = threadIdx.x;
  const int wid = tid >> 5, lane = tid & 31, hl = lane >> 4, n16 = lane & 15;
  const int wm = (wid & 3) * 32, wn = (wid >> 2) * 32;
  v8f acc[2][2];
#pragma unroll
  for (int im = 0; im < 2; ++im)
#pragma unroll
    for (int in = 0; in < 2; ++in) acc[im][in] = vzero();

  for (int kk = 0; kk < C_; kk += 32) {
    __syncthreads();
    if (kk + 32 < C_ && wid == 0)   // prefetch next A tile rows (global_prefetch_b8)
      __builtin_prefetch(xc + ((size_t)(bI * C_ + kk + 32 + lane) * HW_ + hw0), 0, 0);
    // A tile: planar (k-major) -> [m][k]
    for (int i = tid; i < 32 * 64; i += 256) {
      int k = i >> 6, c2 = (i & 63) * 2;
      const bf16* g = xc + ((size_t)(bI * C_ + kk + k) * HW_ + hw0 + c2);
      sA[c2 * 34 + k]       = g[0];
      sA[(c2 + 1) * 34 + k] = g[1];
    }
    // B tile: lin_w[n][k] -> [n][k]
    for (int i = tid; i < 64 * 32; i += 256) {
      int n = i >> 5, k = i & 31;
      sB[n * 34 + k] = (bf16)lw[(size_t)(nbase + n) * C_ + kk + k];
    }
    __syncthreads();
#pragma unroll
    for (int im = 0; im < 2; ++im) {
      v16bf a = frag_a(&sA[(wm + im * 16) * 34], 34, 1);
#pragma unroll
      for (int in = 0; in < 2; ++in) {
        v16bf b = frag_b(&sB[(wn + in * 16) * 34], 1, 34);
        acc[im][in] = wmma_bf16(a, b, acc[im][in]);
      }
    }
  }
  // epilogue: bias, stage in LDS, planar coalesced store
#pragma unroll
  for (int im = 0; im < 2; ++im)
#pragma unroll
    for (int in = 0; in < 2; ++in)
#pragma unroll
      for (int r = 0; r < 8; ++r) {
        int m = wm + im * 16 + r + hl * 8;
        int n = wn + in * 16 + n16;
        sO[n * 130 + m] = (bf16)(acc[im][in][r] + lb[nbase + n]);
      }
  __syncthreads();
  for (int i = tid; i < 64 * 128; i += 256) {
    int n = i >> 7, m = i & 127;
    int ng = nbase + n;
    bf16 v = sO[n * 130 + m];
    if (ng < C_) xout[(size_t)(bI * C_ + ng) * HW_ + hw0 + m] = v;
    else         zout[(size_t)(bI * C_ + ng - C_) * HW_ + hw0 + m] = v;
  }
}

// ================= K4: per-(b,c) plane  cos@P@cos^T * wmod -> cos^T@(.)@cos =================
__global__ __launch_bounds__(256) void k_dct(const bf16* __restrict__ xin,
                                             const bf16* __restrict__ cosg,
                                             const float* __restrict__ wmod,
                                             bf16* __restrict__ yout) {
  const int plane = blockIdx.x;       // b*C + c
  const int c = plane % C_;
  __shared__ __align__(16) bf16 sC[64 * PAD_];
  __shared__ __align__(16) bf16 sP[64 * PAD_];
  __shared__ __align__(16) bf16 sT[64 * PAD_];
  const int tid = threadIdx.x;
  const int wid = tid >> 5, lane = tid & 31, hl = lane >> 4, n16 = lane & 15;
  const bf16* src = xin + (size_t)plane * HW_;
  // async global->LDS copies (8 bf16 = 16B per op, row-aligned), ASYNCcnt path
  for (int i = tid; i < 512; i += 256) {
    int r = i >> 3, c0 = (i & 7) * 8;
    async_copy16(src + r * 64 + c0,  &sP[r * PAD_ + c0]);
    async_copy16(cosg + r * 64 + c0, &sC[r * PAD_ + c0]);
  }
  wait_async0();
  __syncthreads();
  // stage 1: T1[n][w] = sum_h cos[n][h] * P[h][w]
  for (int j = 0; j < 2; ++j) {
    int t = wid * 2 + j, tm = t >> 2, tn = t & 3;
    v8f acc = mm64(sC, PAD_, 1, sP, PAD_, 1, tm, tn);
    store_tile_bf16(sT, PAD_, tm, tn, acc);
  }
  __syncthreads();
  // stage 2: T2[n][m2] = sum_w T1[n][w] * cos[m2][w]; modulate
  for (int j = 0; j < 2; ++j) {
    int t = wid * 2 + j, tm = t >> 2, tn = t & 3;
    v8f acc = mm64(sT, PAD_, 1, sC, 1, PAD_, tm, tn);
#pragma unroll
    for (int r = 0; r < 8; ++r) {
      int fn = tm * 16 + r + hl * 8;
      int fm = tn * 16 + n16;
      acc[r] *= wmod[(size_t)((fn << 6) + fm) * C_ + c];
    }
    store_tile_bf16(sP, PAD_, tm, tn, acc);
  }
  __syncthreads();
  // stage 3: T3[h][m2] = sum_n cos[n][h] * T2[n][m2]
  for (int j = 0; j < 2; ++j) {
    int t = wid * 2 + j, tm = t >> 2, tn = t & 3;
    v8f acc = mm64(sC, 1, PAD_, sP, PAD_, 1, tm, tn);
    store_tile_bf16(sT, PAD_, tm, tn, acc);
  }
  __syncthreads();
  // stage 4: Y[h][w] = sum_m2 T3[h][m2] * cos[m2][w]
  for (int j = 0; j < 2; ++j) {
    int t = wid * 2 + j, tm = t >> 2, tn = t & 3;
    v8f acc = mm64(sT, PAD_, 1, sC, PAD_, 1, tm, tn);
    store_tile_bf16(sP, PAD_, tm, tn, acc);
  }
  __syncthreads();
  bf16* dst = yout + (size_t)plane * HW_;
  for (int i = tid; i < HW_; i += 256) dst[i] = sP[(i >> 6) * PAD_ + (i & 63)];
}

// ================= K5: LayerNorm stats per position =================
__global__ __launch_bounds__(256) void k_stats(const bf16* __restrict__ y,
                                               float2* __restrict__ stats) {
  int p = blockIdx.x * 256 + threadIdx.x;
  int b = p >> 12, hw = p & 4095;
  const bf16* base = y + (size_t)b * C_ * HW_ + hw;
  float s = 0.f, s2 = 0.f;
  for (int cix = 0; cix < C_; ++cix) {
    float v = (float)base[(size_t)cix * HW_];
    s += v; s2 += v * v;
  }
  float mu = s * (1.f / C_);
  float var = s2 * (1.f / C_) - mu * mu;
  float2 st; st.x = mu; st.y = rsqrtf(var + 1e-5f);
  stats[p] = st;
}

// ================= K6: out = (LN(y)*silu(z)) @ out_w^T + out_b -> fp32 planar =================
__global__ __launch_bounds__(256) void k_gemm_out(const bf16* __restrict__ y,
                                                  const bf16* __restrict__ z,
                                                  const float2* __restrict__ stats,
                                                  const float* __restrict__ nw,
                                                  const float* __restrict__ nb,
                                                  const float* __restrict__ ow,
                                                  const float* __restrict__ ob,
                                                  float* __restrict__ out) {
  const int pbase = blockIdx.x * 128;
  const int nbase = blockIdx.y * 64;
  const int bI = pbase >> 12, hw0 = pbase & 4095;
  __shared__ bf16  sA[128 * 34];
  __shared__ bf16  sB[64 * 34];
  __shared__ float sO[64 * 130];
  const int tid = threadIdx.x;
  const int wid = tid >> 5, lane = tid & 31, hl = lane >> 4, n16 = lane & 15;
  const int wm = (wid & 3) * 32, wn = (wid >> 2) * 32;
  v8f acc[2][2];
#pragma unroll
  for (int im = 0; im < 2; ++im)
#pragma unroll
    for (int in = 0; in < 2; ++in) acc[im][in] = vzero();

  for (int kk = 0; kk < C_; kk += 32) {
    __syncthreads();
    if (kk + 32 < C_ && wid == 0)
      __builtin_prefetch(y + ((size_t)(bI * C_ + kk + 32 + lane) * HW_ + hw0), 0, 0);
    // A tile: fused LN + SiLU gate
    for (int i = tid; i < 32 * 128; i += 256) {
      int k = i >> 7, m = i & 127;
      int kg = kk + k;
      size_t off = (size_t)(bI * C_ + kg) * HW_ + hw0 + m;
      float yv = (float)y[off];
      float zv = (float)z[off];
      float2 st = stats[pbase + m];
      float g = zv / (1.f + expf(-zv));
      sA[m * 34 + k] = (bf16)((((yv - st.x) * st.y) * nw[kg] + nb[kg]) * g);
    }
    for (int i = tid; i < 64 * 32; i += 256) {
      int n = i >> 5, k = i & 31;
      sB[n * 34 + k] = (bf16)ow[(size_t)(nbase + n) * C_ + kk + k];
    }
    __syncthreads();
#pragma unroll
    for (int im = 0; im < 2; ++im) {
      v16bf a = frag_a(&sA[(wm + im * 16) * 34], 34, 1);
#pragma unroll
      for (int in = 0; in < 2; ++in) {
        v16bf b = frag_b(&sB[(wn + in * 16) * 34], 1, 34);
        acc[im][in] = wmma_bf16(a, b, acc[im][in]);
      }
    }
  }
#pragma unroll
  for (int im = 0; im < 2; ++im)
#pragma unroll
    for (int in = 0; in < 2; ++in)
#pragma unroll
      for (int r = 0; r < 8; ++r) {
        int m = wm + im * 16 + r + hl * 8;
        int n = wn + in * 16 + n16;
        sO[n * 130 + m] = acc[im][in][r] + ob[nbase + n];
      }
  __syncthreads();
  for (int i = tid; i < 64 * 128; i += 256) {
    int n = i >> 7, m = i & 127;
    out[(size_t)(bI * C_ + nbase + n) * HW_ + hw0 + m] = sO[n * 130 + m];
  }
}

// ================= launch =================
extern "C" void kernel_launch(void* const* d_in, const int* in_sizes, int n_in,
                              void* d_out, int out_size, void* d_ws, size_t ws_size,
                              hipStream_t stream) {
  (void)in_sizes; (void)n_in; (void)out_size; (void)ws_size;
  const float* x    = (const float*)d_in[0];
  const float* fe   = (const float*)d_in[1];
  const float* dwW  = (const float*)d_in[2];
  const float* dwB  = (const float*)d_in[3];
  const float* linw = (const float*)d_in[4];
  const float* linb = (const float*)d_in[5];
  const float* tkw  = (const float*)d_in[6];
  const float* tkb  = (const float*)d_in[7];
  const float* nw   = (const float*)d_in[8];
  const float* nb   = (const float*)d_in[9];
  const float* ow   = (const float*)d_in[10];
  const float* ob   = (const float*)d_in[11];
  const float* cs   = (const float*)d_in[12];
  const float* al   = (const float*)d_in[13];
  float* out = (float*)d_out;

  char* ws = (char*)d_ws;
  bf16*   cosB  = (bf16*)(ws + WS_COS);
  float*  wmod  = (float*)(ws + WS_WMOD);
  float2* stats = (float2*)(ws + WS_STATS);
  bf16*   bufA  = (bf16*)(ws + WS_A);   // xc, later reused for y
  bf16*   bufX  = (bf16*)(ws + WS_X);
  bf16*   bufZ  = (bf16*)(ws + WS_Z);

  k_cos   <<<16, 256, 0, stream>>>(cosB);
  k_wmod  <<<dim3(32, 6), 256, 0, stream>>>(fe, tkw, tkb, cs, al, wmod);
  k_dwconv<<<B_ * C_, 256, 0, stream>>>(x, dwW, dwB, bufA);
  k_gemm_lin<<<dim3(BHW_ / 128, 12), 256, 0, stream>>>(bufA, linw, linb, bufX, bufZ);
  k_dct   <<<B_ * C_, 256, 0, stream>>>(bufX, cosB, wmod, bufA);   // bufA reused as y
  k_stats <<<BHW_ / 256, 256, 0, stream>>>(bufA, stats);
  k_gemm_out<<<dim3(BHW_ / 128, 6), 256, 0, stream>>>(bufA, bufZ, stats, nw, nb, ow, ob, out);
}
